// CausalSelfAttention_85615878079010
// MI455X (gfx1250) — matrix operand order repaired and checked by reading, not compiled, not run on hardware
//
#include <hip/hip_runtime.h>
#include <math.h>

#define B_ 4
#define T_ 2048
#define C_ 768
#define H_ 6
#define D_ 128
#define M_ (B_*T_)

typedef __attribute__((ext_vector_type(16))) __bf16 bf16x16;
typedef __attribute__((ext_vector_type(8)))  float  floatx8;
typedef __attribute__((ext_vector_type(4)))  unsigned int uint4v;
typedef __attribute__((ext_vector_type(8)))  int int8v;
typedef __attribute__((ext_vector_type(4)))  int int4v;

union FragU { uint4 q[2]; bf16x16 v; };

__device__ inline unsigned short f32_to_bf16(float f) {
    union { float f; unsigned u; } u; u.f = f;
    unsigned r = u.u + 0x7FFFu + ((u.u >> 16) & 1u);
    return (unsigned short)(r >> 16);
}
__device__ inline float bf16_to_f32(unsigned short h) {
    union { unsigned u; float f; } u; u.u = ((unsigned)h) << 16;
    return u.f;
}

// A fragment: 16x32 bf16 (M x K). lane = M row (lane&15), half = lane>>4.
__device__ inline bf16x16 load_frag_a(const unsigned short* base, int stride,
                                      int mrow, int kbase, int half) {
    FragU f;
    const unsigned short* p = base + (size_t)mrow * stride + kbase;
    f.q[0] = *(const uint4*)(p + 8 * half);
    f.q[1] = *(const uint4*)(p + 16 + 8 * half);
    return f.v;
}
// B fragment: 32x16 bf16 (K x N), source stored N x K row-major.
__device__ inline bf16x16 load_frag_b(const unsigned short* base, int stride,
                                      int nrow, int kbase, int half) {
    FragU f;
    const unsigned short* p = base + (size_t)nrow * stride + kbase;
    f.q[0] = *(const uint4*)(p + 8 * half);
    f.q[1] = *(const uint4*)(p + 16 + 8 * half);
    return f.v;
}

__device__ inline floatx8 wmma_bf16(bf16x16 a, bf16x16 b, floatx8 c) {
    return __builtin_amdgcn_wmma_f32_16x16x32_bf16(
        false, a, false, b, (short)0, c, false, false);
}

// gfx1250 async global->LDS copy, 16B per lane, tracked by ASYNCcnt.
__device__ inline void async_copy_b128(unsigned lds_off, const void* gptr) {
    unsigned long long ga = (unsigned long long)(uintptr_t)gptr;
    asm volatile("global_load_async_to_lds_b128 %0, %1, off"
                 :: "v"(lds_off), "v"(ga) : "memory");
}
__device__ inline void wait_async_le3() { asm volatile("s_wait_asynccnt 0x3" ::: "memory"); }
__device__ inline void wait_async_0()   { asm volatile("s_wait_asynccnt 0x0" ::: "memory"); }
__device__ inline unsigned lds_offset(const void* p) {
    return (unsigned)(uintptr_t)p;   // low 32 bits of generic addr = LDS byte offset
}

#if __has_builtin(__builtin_amdgcn_tensor_load_to_lds)
#define HAVE_TDM 1
// TDM: load one 32x128 bf16 tile (row stride 128 elems) from global to LDS.
__device__ inline void tdm_load_tile_32x128(unsigned lds_off, const unsigned short* gsrc) {
    unsigned long long ga = (unsigned long long)(uintptr_t)gsrc;
    // D# group0: count=1 | lds_addr | global_addr[56:0] | type=2
    uint4v g0 = { 1u, lds_off, (unsigned)ga,
                  (unsigned)((ga >> 32) & 0x01ffffffu) | 0x80000000u };
    // D# group1: data_size=2B; tensor_dim0=128; tensor_dim1=2048; tile_dim0=128;
    //            tile_dim1=32; tensor_dim0_stride=128
    int8v g1 = { 0x10000,
                 (int)(128u << 16),     // tensor_dim0[15:0] @ bits 63:48
                 (int)(2048u << 16),    // tensor_dim1[15:0] @ bits 95:80
                 (int)(128u << 16),     // tile_dim0 @ bits 127:112
                 32,                    // tile_dim1 (tile_dim2 = 0)
                 128,                   // tensor_dim0_stride[31:0]
                 0, 0 };
    int4v z4 = { 0, 0, 0, 0 };
#if defined(__clang_major__) && __clang_major__ >= 23
    int8v z8 = { 0, 0, 0, 0, 0, 0, 0, 0 };
    __builtin_amdgcn_tensor_load_to_lds(g0, g1, z4, z4, z8, 0);
#else
    __builtin_amdgcn_tensor_load_to_lds(g0, g1, z4, z4, 0);
#endif
}
#else
#define HAVE_TDM 0
#endif

// ---------------------------------------------------------------- cast fp32->bf16
__global__ __launch_bounds__(256) void cast_bf16(const float* __restrict__ in,
                                                 unsigned short* __restrict__ out,
                                                 int n) {
    int i = blockIdx.x * 256 + threadIdx.x;
    if (i < n) out[i] = f32_to_bf16(in[i]);
}

// ---------------------------------------------------------------- GEMM: C = A * W^T
// Block tile 64x128, 8 waves each 32x32. Double-buffered LDS fed by async copies,
// software-pipelined (peeled epilogue, branch-free steady state).
// mode 0: Out = float M x N flat ; mode 1: Out = bf16 scattered to (B,H,T,D)
__global__ __launch_bounds__(256) void gemm_bf16(const unsigned short* __restrict__ A,
                                                 const unsigned short* __restrict__ Wt,
                                                 void* __restrict__ Out,
                                                 int M, int N, int K, int mode) {
    __shared__ __align__(16) unsigned short As[2][64][32];
    __shared__ __align__(16) unsigned short Bs[2][128][32];

    const int tid  = threadIdx.x;
    const int lane = tid & 31;
    const int wid  = tid >> 5;
    const int half = lane >> 4;
    const int nIdx = lane & 15;

    const int m0 = blockIdx.y * 64;
    const int n0 = blockIdx.x * 128;
    const int wave_row = wid & 1;    // 0..1 -> 32-row strip
    const int wave_col = wid >> 1;   // 0..3 -> 32-col strip

    // staging: A = 256 chunks (1/thread), B = 512 chunks (2/thread)
    const int aRow = tid >> 2;
    const int aCh  = (tid & 3) * 8;
    const unsigned short* Ag  = &A[(size_t)(m0 + aRow) * K + aCh];
    const unsigned short* Bg0 = &Wt[(size_t)(n0 + aRow) * K + aCh];
    const unsigned short* Bg1 = &Wt[(size_t)(n0 + 64 + aRow) * K + aCh];

    unsigned ldsA[2], ldsB0[2], ldsB1[2];
    #pragma unroll
    for (int p = 0; p < 2; ++p) {
        ldsA[p]  = lds_offset(&As[p][aRow][aCh]);
        ldsB0[p] = lds_offset(&Bs[p][aRow][aCh]);
        ldsB1[p] = lds_offset(&Bs[p][64 + aRow][aCh]);
    }

    floatx8 acc[2][2];
    #pragma unroll
    for (int i = 0; i < 2; ++i)
        #pragma unroll
        for (int j = 0; j < 2; ++j) acc[i][j] = (floatx8){0,0,0,0,0,0,0,0};

    // prologue: stage tile 0
    async_copy_b128(ldsA[0],  Ag);
    async_copy_b128(ldsB0[0], Bg0);
    async_copy_b128(ldsB1[0], Bg1);

    const int nk = K / 32;
    // steady state: unconditionally stage tile kt+1, wait for tile kt, compute
    for (int kt = 0; kt < nk - 1; ++kt) {
        const int cur = kt & 1;
        const int nxt = cur ^ 1;
        async_copy_b128(ldsA[nxt],  Ag  + (size_t)(kt + 1) * 32);
        async_copy_b128(ldsB0[nxt], Bg0 + (size_t)(kt + 1) * 32);
        async_copy_b128(ldsB1[nxt], Bg1 + (size_t)(kt + 1) * 32);
        wait_async_le3();    // 3 oldest (tile kt) have landed in LDS
        __syncthreads();

        bf16x16 a0 = load_frag_a(&As[cur][0][0], 32, wave_row * 32 + nIdx,      0, half);
        bf16x16 a1 = load_frag_a(&As[cur][0][0], 32, wave_row * 32 + 16 + nIdx, 0, half);
        bf16x16 b0 = load_frag_b(&Bs[cur][0][0], 32, wave_col * 32 + nIdx,      0, half);
        bf16x16 b1 = load_frag_b(&Bs[cur][0][0], 32, wave_col * 32 + 16 + nIdx, 0, half);
        acc[0][0] = wmma_bf16(a0, b0, acc[0][0]);
        acc[0][1] = wmma_bf16(a0, b1, acc[0][1]);
        acc[1][0] = wmma_bf16(a1, b0, acc[1][0]);
        acc[1][1] = wmma_bf16(a1, b1, acc[1][1]);
        __syncthreads();
    }
    // epilogue tile
    {
        const int cur = (nk - 1) & 1;
        wait_async_0();
        __syncthreads();
        bf16x16 a0 = load_frag_a(&As[cur][0][0], 32, wave_row * 32 + nIdx,      0, half);
        bf16x16 a1 = load_frag_a(&As[cur][0][0], 32, wave_row * 32 + 16 + nIdx, 0, half);
        bf16x16 b0 = load_frag_b(&Bs[cur][0][0], 32, wave_col * 32 + nIdx,      0, half);
        bf16x16 b1 = load_frag_b(&Bs[cur][0][0], 32, wave_col * 32 + 16 + nIdx, 0, half);
        acc[0][0] = wmma_bf16(a0, b0, acc[0][0]);
        acc[0][1] = wmma_bf16(a0, b1, acc[0][1]);
        acc[1][0] = wmma_bf16(a1, b0, acc[1][0]);
        acc[1][1] = wmma_bf16(a1, b1, acc[1][1]);
    }

    #pragma unroll
    for (int ar = 0; ar < 2; ++ar) {
        #pragma unroll
        for (int ct = 0; ct < 2; ++ct) {
            #pragma unroll
            for (int v = 0; v < 8; ++v) {
                int m = m0 + wave_row * 32 + ar * 16 + v + 8 * half;
                int n = n0 + wave_col * 32 + ct * 16 + nIdx;
                float val = acc[ar][ct][v];
                if (mode == 0) {
                    ((float*)Out)[(size_t)m * N + n] = val;
                } else {
                    int b = m / T_, t = m % T_;
                    int h = n >> 7, d = n & 127;
                    ((unsigned short*)Out)[((((size_t)b * H_ + h) * T_) + t) * D_ + d] =
                        f32_to_bf16(val);
                }
            }
        }
    }
}

// ---------------------------------------------------------------- RoPE + RMSNorm (in place)
__global__ __launch_bounds__(128) void rope_rms(unsigned short* __restrict__ Q,
                                                unsigned short* __restrict__ Kk,
                                                const float* __restrict__ cosb,
                                                const float* __restrict__ sinb) {
    const int lane = threadIdx.x & 31;
    const int wid  = threadIdx.x >> 5;
    const int row  = blockIdx.x * 4 + wid;       // 0 .. B*H*T-1
    const int t    = row % T_;
    unsigned short* P = (blockIdx.y == 0) ? Q : Kk;
    P += (size_t)row * D_;

    const int d0 = lane, d1 = lane + 32;
    float c0 = cosb[t * 64 + d0], s0 = sinb[t * 64 + d0];
    float c1 = cosb[t * 64 + d1], s1 = sinb[t * 64 + d1];

    float x1a = bf16_to_f32(P[d0]), x2a = bf16_to_f32(P[d0 + 64]);
    float x1b = bf16_to_f32(P[d1]), x2b = bf16_to_f32(P[d1 + 64]);
    float y1a =  x1a * c0 + x2a * s0;
    float y2a = -x1a * s0 + x2a * c0;
    float y1b =  x1b * c1 + x2b * s1;
    float y2b = -x1b * s1 + x2b * c1;

    float ss = y1a * y1a + y2a * y2a + y1b * y1b + y2b * y2b;
    #pragma unroll
    for (int m = 1; m < 32; m <<= 1) ss += __shfl_xor(ss, m, 32);
    float scale = 1.0f / sqrtf(ss * (1.0f / 128.0f) + 1e-15f);

    P[d0]      = f32_to_bf16(y1a * scale);
    P[d0 + 64] = f32_to_bf16(y2a * scale);
    P[d1]      = f32_to_bf16(y1b * scale);
    P[d1 + 64] = f32_to_bf16(y2b * scale);
}

// ---------------------------------------------------------------- Flash attention
__global__ __launch_bounds__(128) void attn_fwd(const unsigned short* __restrict__ Qh,
                                                const unsigned short* __restrict__ Kh,
                                                const unsigned short* __restrict__ Vh,
                                                unsigned short* __restrict__ Ob) {
    __shared__ __align__(16) unsigned short Ks[32][128];   // [kv][d]
    __shared__ __align__(16) unsigned short Vt[128][32];   // [d][kv] transposed
    __shared__ __align__(16) unsigned short Ps[4][16][32]; // per-wave P staging

    const int tid  = threadIdx.x;
    const int lane = tid & 31;
    const int wid  = tid >> 5;
    const int half = lane >> 4;
    const int nIdx = lane & 15;

    const int bh = blockIdx.y;
    const int b  = bh / H_;
    const int h  = bh % H_;
    const int qblock = blockIdx.x * 64;
    const int q0     = qblock + wid * 16;

    const unsigned short* Qp = Qh + (size_t)bh * T_ * D_;
    const unsigned short* Kp = Kh + (size_t)bh * T_ * D_;
    const unsigned short* Vp = Vh + (size_t)bh * T_ * D_;

    bf16x16 qf[4];
    #pragma unroll
    for (int ks = 0; ks < 4; ++ks)
        qf[ks] = load_frag_a(Qp, D_, q0 + nIdx, ks * 32, half);

    float m_i[8], l_i[8];
    floatx8 O[8];
    #pragma unroll
    for (int v = 0; v < 8; ++v) { m_i[v] = -INFINITY; l_i[v] = 0.0f; }
    #pragma unroll
    for (int dt = 0; dt < 8; ++dt) O[dt] = (floatx8){0,0,0,0,0,0,0,0};

    const float sm_scale = 0.08838834764831845f;
    const int kv_end = qblock + 64;
    const unsigned ksOff = lds_offset(&Ks[0][0]);

    for (int kv0 = 0; kv0 < kv_end; kv0 += 32) {
        __syncthreads();
#if HAVE_TDM
        if (tid < 32) {   // wave 0 drives the Tensor Data Mover for the K tile
            tdm_load_tile_32x128(ksOff, Kp + (size_t)kv0 * D_);
            __builtin_amdgcn_s_wait_tensorcnt(0);
        }
#else
        #pragma unroll
        for (int i = 0; i < 4; ++i) {
            int c = tid + i * 128, row = c >> 4, ch = (c & 15) * 8;
            *(uint4*)&Ks[row][ch] = *(const uint4*)&Kp[(size_t)(kv0 + row) * D_ + ch];
        }
#endif
        // V tile staged transposed (TDM cannot transpose)
        #pragma unroll
        for (int i = 0; i < 4; ++i) {
            int c = tid + i * 128, row = c >> 4, ch = (c & 15) * 8;
            uint4 vq = *(const uint4*)&Vp[(size_t)(kv0 + row) * D_ + ch];
            const unsigned short* pv = (const unsigned short*)&vq;
            #pragma unroll
            for (int j = 0; j < 8; ++j) Vt[ch + j][row] = pv[j];
        }
        if (kv0 + 32 < kv_end) {   // gfx1250 global_prefetch_b8
            __builtin_prefetch(&Kp[(size_t)(kv0 + 32 + (tid >> 2)) * D_], 0, 1);
            __builtin_prefetch(&Vp[(size_t)(kv0 + 32 + (tid >> 2)) * D_], 0, 1);
        }
        __syncthreads();

        // S = Q @ K^T, two 16x16 column tiles
        floatx8 a0 = {0,0,0,0,0,0,0,0};
        floatx8 a1 = {0,0,0,0,0,0,0,0};
        #pragma unroll
        for (int ks = 0; ks < 4; ++ks) {
            bf16x16 bk0 = load_frag_b(&Ks[0][0], 128, nIdx,      ks * 32, half);
            bf16x16 bk1 = load_frag_b(&Ks[0][0], 128, 16 + nIdx, ks * 32, half);
            a0 = wmma_bf16(qf[ks], bk0, a0);
            a1 = wmma_bf16(qf[ks], bk1, a1);
        }

        // online softmax: row m = v + 8*half, col n = nIdx
        float alp[8];
        #pragma unroll
        for (int v = 0; v < 8; ++v) {
            int qg = q0 + v + 8 * half;
            float s0 = a0[v] * sm_scale;
            float s1 = a1[v] * sm_scale;
            if (kv0 + nIdx      > qg) s0 = -INFINITY;
            if (kv0 + 16 + nIdx > qg) s1 = -INFINITY;
            float r = fmaxf(s0, s1);
            #pragma unroll
            for (int m = 1; m < 16; m <<= 1) r = fmaxf(r, __shfl_xor(r, m, 32));
            float mnew  = fmaxf(m_i[v], r);
            float alpha = __expf(m_i[v] - mnew);
            float p0 = __expf(s0 - mnew);
            float p1 = __expf(s1 - mnew);
            float rs = p0 + p1;
            #pragma unroll
            for (int m = 1; m < 16; m <<= 1) rs += __shfl_xor(rs, m, 32);
            l_i[v] = l_i[v] * alpha + rs;
            m_i[v] = mnew;
            alp[v] = alpha;
            Ps[wid][v + 8 * half][nIdx]      = f32_to_bf16(p0);
            Ps[wid][v + 8 * half][16 + nIdx] = f32_to_bf16(p1);
        }
        #pragma unroll
        for (int dt = 0; dt < 8; ++dt)
            #pragma unroll
            for (int v = 0; v < 8; ++v) O[dt][v] *= alp[v];

        // P (16x32) @ V (32x128)
        bf16x16 pa = load_frag_a(&Ps[wid][0][0], 32, nIdx, 0, half);
        #pragma unroll
        for (int dt = 0; dt < 8; ++dt) {
            bf16x16 bv = load_frag_b(&Vt[0][0], 32, dt * 16 + nIdx, 0, half);
            O[dt] = wmma_bf16(pa, bv, O[dt]);
        }
    }

    #pragma unroll
    for (int dt = 0; dt < 8; ++dt) {
        #pragma unroll
        for (int v = 0; v < 8; ++v) {
            int qg = q0 + v + 8 * half;
            int d  = dt * 16 + nIdx;
            float val = O[dt][v] / l_i[v];
            Ob[((size_t)b * T_ + qg) * C_ + h * D_ + d] = f32_to_bf16(val);
        }
    }
}

// ---------------------------------------------------------------- launch
extern "C" void kernel_launch(void* const* d_in, const int* in_sizes, int n_in,
                              void* d_out, int out_size, void* d_ws, size_t ws_size,
                              hipStream_t stream) {
    const float* x    = (const float*)d_in[0];
    const float* cosb = (const float*)d_in[1];
    const float* sinb = (const float*)d_in[2];
    const float* Wq   = (const float*)d_in[3];
    const float* Wk   = (const float*)d_in[4];
    const float* Wv   = (const float*)d_in[5];
    const float* Wo   = (const float*)d_in[6];

    char* ws = (char*)d_ws;
    unsigned short* xb  = (unsigned short*)ws; ws += (size_t)M_ * C_ * 2;
    unsigned short* Wqb = (unsigned short*)ws; ws += (size_t)C_ * C_ * 2;
    unsigned short* Wkb = (unsigned short*)ws; ws += (size_t)C_ * C_ * 2;
    unsigned short* Wvb = (unsigned short*)ws; ws += (size_t)C_ * C_ * 2;
    unsigned short* Wob = (unsigned short*)ws; ws += (size_t)C_ * C_ * 2;
    unsigned short* Qh  = (unsigned short*)ws; ws += (size_t)M_ * C_ * 2;
    unsigned short* Kh  = (unsigned short*)ws; ws += (size_t)M_ * C_ * 2;
    unsigned short* Vh  = (unsigned short*)ws; ws += (size_t)M_ * C_ * 2;
    unsigned short* Ab  = (unsigned short*)ws; ws += (size_t)M_ * C_ * 2;

    const int nx = M_ * C_;
    const int nw = C_ * C_;
    cast_bf16<<<(nx + 255) / 256, 256, 0, stream>>>(x,  xb,  nx);
    cast_bf16<<<(nw + 255) / 256, 256, 0, stream>>>(Wq, Wqb, nw);
    cast_bf16<<<(nw + 255) / 256, 256, 0, stream>>>(Wk, Wkb, nw);
    cast_bf16<<<(nw + 255) / 256, 256, 0, stream>>>(Wv, Wvb, nw);
    cast_bf16<<<(nw + 255) / 256, 256, 0, stream>>>(Wo, Wob, nw);

    dim3 ggrid(C_ / 128, M_ / 64);
    gemm_bf16<<<ggrid, 256, 0, stream>>>(xb, Wqb, Qh, M_, C_, C_, 1);
    gemm_bf16<<<ggrid, 256, 0, stream>>>(xb, Wkb, Kh, M_, C_, C_, 1);
    gemm_bf16<<<ggrid, 256, 0, stream>>>(xb, Wvb, Vh, M_, C_, C_, 1);

    rope_rms<<<dim3(B_ * H_ * T_ / 4, 2), 128, 0, stream>>>(Qh, Kh, cosb, sinb);

    attn_fwd<<<dim3(T_ / 64, B_ * H_), 128, 0, stream>>>(Qh, Kh, Vh, Ab);

    gemm_bf16<<<ggrid, 256, 0, stream>>>(Ab, Wob, d_out, M_, C_, C_, 0);
}
